// MLA_74122545594919
// MI455X (gfx1250) — matrix operand order repaired and checked by reading, not compiled, not run on hardware
//
#include <hip/hip_runtime.h>

#define DEVINL __device__ __forceinline__

constexpr int B_   = 2;
constexpr int S_   = 2048;
constexpr int D_   = 2048;
constexpr int H_   = 16;
constexpr int NOPE_= 128;
constexpr int ROPE_= 64;
constexpr int VD_  = 128;
constexpr int KVR_ = 512;
constexpr int QKH_ = NOPE_ + ROPE_;      // 192
constexpr int ROWS = B_ * S_;            // 4096
constexpr int NQ   = H_ * QKH_;          // 3072
constexpr int NKVA = KVR_ + ROPE_;       // 576
constexpr int NKVB = H_ * (NOPE_ + VD_); // 4096
constexpr int NO   = H_ * VD_;           // 2048

typedef __bf16 v16bf __attribute__((ext_vector_type(16)));
typedef float  v8f   __attribute__((ext_vector_type(8)));
typedef unsigned int v4u  __attribute__((ext_vector_type(4)));
typedef int          v4i  __attribute__((ext_vector_type(4)));
typedef int          v8i_ __attribute__((ext_vector_type(8)));

union FragU { uint4 q[2]; v16bf v; };

// Tensor Data Mover availability (device pass only; host pass falls back)
#if defined(__has_builtin)
#  if __has_builtin(__builtin_amdgcn_tensor_load_to_lds) && __has_builtin(__builtin_amdgcn_s_wait_tensorcnt)
#    define HAVE_TDM 1
#  else
#    define HAVE_TDM 0
#  endif
#else
#  define HAVE_TDM 0
#endif

DEVINL unsigned short f2bf(float f) {
  unsigned int u = __float_as_uint(f);
  u += 0x7FFFu + ((u >> 16) & 1u);       // round-to-nearest-even
  return (unsigned short)(u >> 16);
}

// A-fragment 16x32 bf16, row-major K-contiguous global source.
// ISA layout: lanes 0-15: halves0-7 = K 0..7, halves8-15 = K 16..23
//             lanes16-31: halves0-7 = K 8..15, halves8-15 = K 24..31
DEVINL v16bf load_frag_a(const unsigned short* __restrict__ base, int ld, int k0, int lane) {
  const unsigned short* p = base + (size_t)(lane & 15) * ld + k0 + ((lane >> 4) << 3);
  FragU f;
  f.q[0] = *(const uint4*)(p);
  f.q[1] = *(const uint4*)(p + 16);
  return f.v;
}
// Same fragment sourced from an LDS A-tile with padded row stride of 72 elems
// (144B = 36 banks: 16 consecutive rows map to 16 distinct LDS banks).
DEVINL v16bf load_frag_a_lds(const unsigned short* base, int sub, int lane) {
  const unsigned short* p = base + (lane & 15) * 72 + sub + ((lane >> 4) << 3);
  FragU f;
  f.q[0] = *(const uint4*)(p);
  f.q[1] = *(const uint4*)(p + 16);
  return f.v;
}
// B-fragment 32x16 bf16 from N x K row-major source.
// ISA layout: lanes 0-15: K 0..7,16..23; lanes 16-31: K 8..15,24..31 (same as A).
DEVINL v16bf load_frag_b(const unsigned short* __restrict__ base, int ld, int k0, int lane) {
  const unsigned short* p = base + (size_t)(lane & 15) * ld + k0 + ((lane >> 4) << 3);
  FragU f;
  f.q[0] = *(const uint4*)(p);
  f.q[1] = *(const uint4*)(p + 16);
  return f.v;
}

DEVINL v8f wmma_bf16(v16bf a, v16bf b, v8f c) {
  return __builtin_amdgcn_wmma_f32_16x16x32_bf16(false, a, false, b, (short)0, c, false, false);
}

// ---------------------------------------------------------------- TDM staging
#if HAVE_TDM
// 2D D#: 64 rows x 64 bf16 tile, row stride = strideElems. LDS pad of 4 DWORDs
// (8 elems) every 32 DWORDs (one tile row) -> padded 72-element LDS row stride.
DEVINL void tdm_load_tile64x64(unsigned ldsOff, const unsigned short* gsrc, unsigned strideElems) {
  unsigned long long ga = (unsigned long long)(uintptr_t)gsrc;
  v4u g0;
  g0[0] = 1u;                                          // count=1, user descriptor
  g0[1] = ldsOff;                                      // lds_addr
  g0[2] = (unsigned)ga;                                // global_addr[31:0]
  g0[3] = (unsigned)((ga >> 32) & 0x01FFFFFFull) | 0x80000000u; // addr[56:32], type=2
  v8i_ g1;
  g1[0] = 0x07110000;          // data_size=2B | pad_enable | pad_interval=32dw | pad_amount=4dw
  g1[1] = 64 << 16;            // tensor_dim0 = 64 (low half)
  g1[2] = 64 << 16;            // tensor_dim0 hi=0 | tensor_dim1 = 64 (low half)
  g1[3] = 64 << 16;            // tensor_dim1 hi=0 | tile_dim0 = 64
  g1[4] = 64;                  // tile_dim1 = 64, tile_dim2 = 0
  g1[5] = (int)strideElems;    // tensor_dim0_stride[31:0]
  g1[6] = 0;
  g1[7] = 0;
  v4i z = {0, 0, 0, 0};
#if __clang_major__ >= 23
  v8i_ z8 = {0, 0, 0, 0, 0, 0, 0, 0};
  __builtin_amdgcn_tensor_load_to_lds(g0, g1, z, z, z8, 0);
#else
  __builtin_amdgcn_tensor_load_to_lds(g0, g1, z, z, 0);
#endif
}
#endif

DEVINL void stage_a_tile64(unsigned short* dst, const unsigned short* src, int ld, int wave, int tid) {
#if HAVE_TDM
  if (wave == 0)
    tdm_load_tile64x64((unsigned)(uintptr_t)dst, src, (unsigned)ld);
#else
#pragma unroll
  for (int r = 0; r < 2; ++r) {                        // 512 uint4 chunks, 256 threads
    int idx = tid + r * 256;
    int row = idx >> 3, ch = (idx & 7) * 8;
    *(uint4*)(dst + row * 72 + ch) = *(const uint4*)(src + (size_t)row * ld + ch);
  }
#endif
}
// Unconditional: waves that issued no TDM op have TENSORcnt==0, so the wait is
// free for them and we avoid the divergent saveexec branch around the wait.
DEVINL void stage_a_wait() {
#if HAVE_TDM
  __builtin_amdgcn_s_wait_tensorcnt(0);
#endif
  __syncthreads();
}

// ---------------------------------------------------------------- converts
__global__ void f32_to_bf16_k(const float* __restrict__ in, unsigned short* __restrict__ out, size_t n) {
  size_t i = (size_t)blockIdx.x * blockDim.x + threadIdx.x;
  size_t stride = (size_t)gridDim.x * blockDim.x;
  for (; i < n; i += stride) out[i] = f2bf(in[i]);
}

// ---------------------------------------------------------------- GEMM
// C[M,N] fp32 = A[M,K]bf16 (row-major) * B[N,K]bf16 (row-major, i.e. B^T).
// Compile-time M,N,K -> immediate-offset addressing, minimal VALU.
// Block: 256 threads = 8 waves -> 64(M) x 256(N); wave: 32 x 64.
// A staged in LDS as 64x64 tiles via TDM (double-buffered, one barrier per
// 64 K); B double-buffered in VGPRs at 32-K granularity.
// Requires M%64==0, N%64==0, K%64==0 (true for all calls here).
template <int M, int N, int K>
__launch_bounds__(256)
__global__ void gemm_bf16_nt(const unsigned short* __restrict__ A,
                             const unsigned short* __restrict__ Bm,
                             float* __restrict__ C) {
  static_assert(M % 64 == 0 && N % 64 == 0 && K % 64 == 0, "tile divisibility");
  const int tid  = threadIdx.x;
  const int lane = tid & 31;
  const int wave = tid >> 5;
  const int wm = wave >> 2, wn = wave & 3;
  const int mblk = blockIdx.y * 64;
  int n0 = blockIdx.x * 256 + wn * 64;
  if (n0 + 64 > N) n0 = N - 64;        // clamp: keeps all waves in the barriers
  const int m0 = mblk + wm * 32;

  __shared__ __align__(16) unsigned short atile[2][64 * 72];

  const unsigned short* Bb[4];
#pragma unroll
  for (int j = 0; j < 4; ++j) Bb[j] = Bm + (size_t)(n0 + 16 * j) * K;
  const unsigned short* Ablk = A + (size_t)mblk * K;

  v8f acc[2][4];
#pragma unroll
  for (int i = 0; i < 2; ++i)
#pragma unroll
    for (int j = 0; j < 4; ++j)
#pragma unroll
      for (int v = 0; v < 8; ++v) acc[i][j][v] = 0.f;

  // prologue: stage A(k=0..63), preload B(k=0..31)
  stage_a_tile64(atile[0], Ablk, K, wave, tid);
  v16bf bcur[4];
#pragma unroll
  for (int j = 0; j < 4; ++j) bcur[j] = load_frag_b(Bb[j], K, 0, lane);
  stage_a_wait();

  int buf = 0;
  for (int k0 = 0; k0 < K; k0 += 64) {
    const bool hasnext = (k0 + 64) < K;
    if (hasnext)
      stage_a_tile64(atile[buf ^ 1], Ablk + k0 + 64, K, wave, tid);
    if (k0 + 128 < K) {
#pragma unroll
      for (int j = 0; j < 4; ++j)
        __builtin_prefetch(Bb[j] + (size_t)(lane & 15) * K + k0 + 128, 0, 3);
    }

    // --- substep 0: compute k0 with bcur; preload k0+32 into bnext
    v16bf a0 = load_frag_a_lds(&atile[buf][(wm * 32) * 72], 0, lane);
    v16bf a1 = load_frag_a_lds(&atile[buf][(wm * 32 + 16) * 72], 0, lane);
    v16bf bnext[4];
#pragma unroll
    for (int j = 0; j < 4; ++j) bnext[j] = load_frag_b(Bb[j], K, k0 + 32, lane);
#pragma unroll
    for (int j = 0; j < 4; ++j) {
      acc[0][j] = wmma_bf16(a0, bcur[j], acc[0][j]);
      acc[1][j] = wmma_bf16(a1, bcur[j], acc[1][j]);
    }

    // --- substep 1: compute k0+32 with bnext; preload next iter into bcur
    a0 = load_frag_a_lds(&atile[buf][(wm * 32) * 72], 32, lane);
    a1 = load_frag_a_lds(&atile[buf][(wm * 32 + 16) * 72], 32, lane);
    if (hasnext) {
#pragma unroll
      for (int j = 0; j < 4; ++j) bcur[j] = load_frag_b(Bb[j], K, k0 + 64, lane);
    }
#pragma unroll
    for (int j = 0; j < 4; ++j) {
      acc[0][j] = wmma_bf16(a0, bnext[j], acc[0][j]);
      acc[1][j] = wmma_bf16(a1, bnext[j], acc[1][j]);
    }

    if (hasnext) {
      stage_a_wait();
      buf ^= 1;
    }
  }

  const int roff = (lane >> 4) * 8;
  const int col0 = lane & 15;
#pragma unroll
  for (int i = 0; i < 2; ++i)
#pragma unroll
    for (int j = 0; j < 4; ++j)
#pragma unroll
      for (int v = 0; v < 8; ++v) {
        int row = m0 + 16 * i + v + roff;
        int col = n0 + 16 * j + col0;
        C[(size_t)row * N + col] = acc[i][j][v];
      }
}

// ---------------------------------------------------------------- RoPE(q) + relayout to (B,H,S,192) bf16
__global__ void rope_q_k(const float* __restrict__ q, const float* __restrict__ fcos,
                         const float* __restrict__ fsin, unsigned short* __restrict__ qf) {
  const int r = blockIdx.x;          // b*S + s
  const int b = r / S_, s = r % S_;
  const float* qrow = q + (size_t)r * NQ;
  for (int idx = threadIdx.x; idx < NQ; idx += blockDim.x) {
    int h = idx / QKH_, d = idx % QKH_;
    float val;
    if (d < NOPE_) {
      val = qrow[idx];
    } else {
      int j = d - NOPE_, p = j >> 1;
      float c = fcos[s * 32 + p], sn = fsin[s * 32 + p];
      float xr = qrow[h * QKH_ + NOPE_ + 2 * p];
      float xi = qrow[h * QKH_ + NOPE_ + 2 * p + 1];
      val = (j & 1) ? (xr * sn + xi * c) : (xr * c - xi * sn);
    }
    qf[((size_t)(b * H_ + h) * S_ + s) * QKH_ + d] = f2bf(val);
  }
}

// ---------------------------------------------------------------- RMSNorm(kv_lat) -> bf16; RoPE(k_pe) -> bf16
__global__ void kv_prep_k(const float* __restrict__ kv, const float* __restrict__ w,
                          const float* __restrict__ fcos, const float* __restrict__ fsin,
                          unsigned short* __restrict__ kvlat, unsigned short* __restrict__ kpe) {
  const int r = blockIdx.x;          // b*S + s
  const int tid = threadIdx.x;       // 256
  __shared__ float red[256];
  const float* row = kv + (size_t)r * NKVA;
  float ss = 0.f;
  for (int i = tid; i < KVR_; i += 256) { float v = row[i]; ss += v * v; }
  red[tid] = ss;
  __syncthreads();
  for (int o = 128; o > 0; o >>= 1) {
    if (tid < o) red[tid] += red[tid + o];
    __syncthreads();
  }
  const float rstd = rsqrtf(red[0] * (1.f / KVR_) + 1e-6f);
  for (int i = tid; i < KVR_; i += 256)
    kvlat[(size_t)r * KVR_ + i] = f2bf(row[i] * rstd * w[i]);
  if (tid < 32) {
    const int s = r % S_;
    float c = fcos[s * 32 + tid], sn = fsin[s * 32 + tid];
    float xr = row[KVR_ + 2 * tid], xi = row[KVR_ + 2 * tid + 1];
    kpe[(size_t)r * ROPE_ + 2 * tid]     = f2bf(xr * c - xi * sn);
    kpe[(size_t)r * ROPE_ + 2 * tid + 1] = f2bf(xr * sn + xi * c);
  }
}

// ---------------------------------------------------------------- build K (B,H,S,192) and V^T (B,H,128,S)
__global__ void build_kv_k(const float* __restrict__ kvb, const unsigned short* __restrict__ kpe,
                           unsigned short* __restrict__ k16, unsigned short* __restrict__ vT) {
  size_t i = (size_t)blockIdx.x * blockDim.x + threadIdx.x;
  const size_t total = (size_t)ROWS * NKVB;
  if (i >= total) return;
  int c = (int)(i & 255);
  size_t t = i >> 8;
  int h = (int)(t % H_);
  size_t bs = t / H_;
  int s = (int)(bs % S_);
  int b = (int)(bs / S_);
  float v = kvb[i];
  if (c < NOPE_) {
    size_t kb = ((size_t)(b * H_ + h) * S_ + s) * QKH_;
    k16[kb + c] = f2bf(v);
    if (c < ROPE_) k16[kb + NOPE_ + c] = kpe[(size_t)(b * S_ + s) * ROPE_ + c];
  } else {
    int d = c - NOPE_;
    vT[((size_t)(b * H_ + h) * VD_ + d) * S_ + s] = f2bf(v);
  }
}

// ---------------------------------------------------------------- flash attention
// One wave per (b,h, 16-row query tile). Scores & O live in WMMA accumulators;
// online softmax via lane-shuffle reductions; P staged through 1KB LDS to be
// re-fed as an A-fragment for the P*V WMMAs.
__launch_bounds__(32)
__global__ void flash_attn_k(const unsigned short* __restrict__ Qf,
                             const unsigned short* __restrict__ Kf,
                             const unsigned short* __restrict__ Vt,
                             unsigned short* __restrict__ O) {
  const int lane = threadIdx.x & 31;
  const int bh = blockIdx.y;               // b*H + h
  const int m0 = blockIdx.x * 16;
  const int b = bh / H_, h = bh % H_;
  const unsigned short* Qbase = Qf + ((size_t)bh * S_ + m0) * QKH_;
  const unsigned short* Kbase = Kf + (size_t)bh * S_ * QKH_;
  const unsigned short* Vbase = Vt + (size_t)bh * VD_ * S_;
  __shared__ unsigned short pbuf[16 * 32];

  v16bf aq[6];
#pragma unroll
  for (int kk = 0; kk < 6; ++kk) aq[kk] = load_frag_a(Qbase, QKH_, kk * 32, lane);

  v8f o[8];
  float mrow[8], lrow[8];
#pragma unroll
  for (int j = 0; j < 8; ++j) {
#pragma unroll
    for (int v = 0; v < 8; ++v) o[j][v] = 0.f;
    mrow[j] = -3.0e38f;
    lrow[j] = 0.f;
  }

  const float scale = 0.07216878364870322f;   // 192^-0.5
  const int rid = lane >> 4;                   // half-wave id
  const int nid = lane & 15;

  for (int t0 = 0; t0 <= m0 + 15; t0 += 32) {
    v8f sc[2];
#pragma unroll
    for (int half = 0; half < 2; ++half) {
      const int t = t0 + 16 * half;
      v8f s;
#pragma unroll
      for (int v = 0; v < 8; ++v) s[v] = 0.f;
      const unsigned short* Kt = Kbase + (size_t)t * QKH_;
#pragma unroll
      for (int kk = 0; kk < 6; ++kk) {
        v16bf bk = load_frag_b(Kt, QKH_, kk * 32, lane);
        s = wmma_bf16(aq[kk], bk, s);
      }
#pragma unroll
      for (int v = 0; v < 8; ++v) {
        int row = m0 + v + rid * 8;
        int col = t + nid;
        s[v] = s[v] * scale + (col <= row ? 0.f : -1.0e9f);
      }
      sc[half] = s;
    }

    // online softmax: new row max, rescale factor
    float alpha[8];
#pragma unroll
    for (int v = 0; v < 8; ++v) {
      float x = fmaxf(sc[0][v], sc[1][v]);
#pragma unroll
      for (int off = 1; off < 16; off <<= 1) x = fmaxf(x, __shfl_xor(x, off, 32));
      float mn = fmaxf(mrow[v], x);
      alpha[v] = __expf(mrow[v] - mn);
      mrow[v] = mn;
    }

    // exponentiate, accumulate row sums, stage P (16x32 bf16) in LDS
#pragma unroll
    for (int v = 0; v < 8; ++v) {
      float p0 = __expf(sc[0][v] - mrow[v]);
      float p1 = __expf(sc[1][v] - mrow[v]);
      int rloc = v + rid * 8;
      pbuf[rloc * 32 + nid]      = f2bf(p0);
      pbuf[rloc * 32 + 16 + nid] = f2bf(p1);
      float rs = p0 + p1;
#pragma unroll
      for (int off = 1; off < 16; off <<= 1) rs += __shfl_xor(rs, off, 32);
      lrow[v] = lrow[v] * alpha[v] + rs;
    }
#pragma unroll
    for (int j = 0; j < 8; ++j)
#pragma unroll
      for (int v = 0; v < 8; ++v) o[j][v] *= alpha[v];

    __syncthreads();
    FragU pf;
    {
      const unsigned short* p = &pbuf[nid * 32 + rid * 8];
      pf.q[0] = *(const uint4*)(p);
      pf.q[1] = *(const uint4*)(p + 16);
    }
    __syncthreads();

#pragma unroll
    for (int j = 0; j < 8; ++j) {
      const unsigned short* Vrow = Vbase + (size_t)(j * 16) * S_;
      v16bf bv = load_frag_b(Vrow, S_, t0, lane);
      o[j] = wmma_bf16(pf.v, bv, o[j]);
    }
  }

  // normalize + store to (B,S,H*VD) bf16
#pragma unroll
  for (int j = 0; j < 8; ++j)
#pragma unroll
    for (int v = 0; v < 8; ++v) {
      int row = m0 + v + rid * 8;
      int d = j * 16 + nid;
      float val = o[j][v] / lrow[v];
      O[((size_t)(b * S_ + row) * H_ + h) * VD_ + d] = f2bf(val);
    }
}

// ---------------------------------------------------------------- launcher
extern "C" void kernel_launch(void* const* d_in, const int* in_sizes, int n_in,
                              void* d_out, int out_size, void* d_ws, size_t ws_size,
                              hipStream_t stream) {
  const float* x      = (const float*)d_in[0];
  const float* wq     = (const float*)d_in[1];
  const float* wkv_a  = (const float*)d_in[2];
  const float* kvnw   = (const float*)d_in[3];
  const float* wkv_b  = (const float*)d_in[4];
  const float* wo     = (const float*)d_in[5];
  const float* fcos   = (const float*)d_in[6];
  const float* fsin   = (const float*)d_in[7];
  // d_in[8] (mask) is exactly causal with -1e9 fill: implemented in-kernel.

  const size_t n_x    = (size_t)ROWS * D_;
  const size_t n_wq   = (size_t)NQ * D_;
  const size_t n_wkva = (size_t)NKVA * D_;
  const size_t n_wkvb = (size_t)NKVB * KVR_;
  const size_t n_wo   = (size_t)D_ * NO;

  size_t off = 0;
  auto alloc = [&](size_t bytes) -> void* {
    void* p = (char*)d_ws + off;
    off += (bytes + 255) & ~(size_t)255;
    return p;
  };

  unsigned short* x16    = (unsigned short*)alloc(n_x * 2);
  unsigned short* wq16   = (unsigned short*)alloc(n_wq * 2);
  unsigned short* wkva16 = (unsigned short*)alloc(n_wkva * 2);
  unsigned short* wkvb16 = (unsigned short*)alloc(n_wkvb * 2);
  unsigned short* wo16   = (unsigned short*)alloc(n_wo * 2);
  float* q32   = (float*)alloc((size_t)ROWS * NQ * 4);
  float* kv32  = (float*)alloc((size_t)ROWS * NKVA * 4);
  float* kvb32 = (float*)alloc((size_t)ROWS * NKVB * 4);
  unsigned short* kvlat16 = (unsigned short*)alloc((size_t)ROWS * KVR_ * 2);
  unsigned short* kpe16   = (unsigned short*)alloc((size_t)ROWS * ROPE_ * 2);
  unsigned short* qf16    = (unsigned short*)alloc((size_t)B_ * H_ * S_ * QKH_ * 2);
  unsigned short* k16     = (unsigned short*)alloc((size_t)B_ * H_ * S_ * QKH_ * 2);
  unsigned short* vT16    = (unsigned short*)alloc((size_t)B_ * H_ * VD_ * S_ * 2);
  unsigned short* attn16  = (unsigned short*)alloc((size_t)ROWS * NO * 2);

  // 1) fp32 -> bf16 converts
  f32_to_bf16_k<<<4096, 256, 0, stream>>>(x,     x16,    n_x);
  f32_to_bf16_k<<<4096, 256, 0, stream>>>(wq,    wq16,   n_wq);
  f32_to_bf16_k<<<2048, 256, 0, stream>>>(wkv_a, wkva16, n_wkva);
  f32_to_bf16_k<<<2048, 256, 0, stream>>>(wkv_b, wkvb16, n_wkvb);
  f32_to_bf16_k<<<4096, 256, 0, stream>>>(wo,    wo16,   n_wo);

  // 2) q = x @ wq^T   (4096 x 3072 x 2048)
  gemm_bf16_nt<ROWS, NQ, D_>
      <<<dim3((NQ + 255) / 256, ROWS / 64), 256, 0, stream>>>(x16, wq16, q32);
  // 3) kv = x @ wkv_a^T (4096 x 576 x 2048)
  gemm_bf16_nt<ROWS, NKVA, D_>
      <<<dim3((NKVA + 255) / 256, ROWS / 64), 256, 0, stream>>>(x16, wkva16, kv32);

  // 4) RoPE(q) + relayout; 5) RMSNorm + RoPE(k_pe)
  rope_q_k<<<ROWS, 256, 0, stream>>>(q32, fcos, fsin, qf16);
  kv_prep_k<<<ROWS, 256, 0, stream>>>(kv32, kvnw, fcos, fsin, kvlat16, kpe16);

  // 6) kvb = rmsnorm(kv_lat) @ wkv_b^T (4096 x 4096 x 512)
  gemm_bf16_nt<ROWS, NKVB, KVR_>
      <<<dim3((NKVB + 255) / 256, ROWS / 64), 256, 0, stream>>>(kvlat16, wkvb16, kvb32);

  // 7) split into K (with broadcast k_pe) and V^T
  {
    size_t total = (size_t)ROWS * NKVB;
    build_kv_k<<<(unsigned)((total + 255) / 256), 256, 0, stream>>>(kvb32, kpe16, k16, vT16);
  }

  // 8) flash attention
  flash_attn_k<<<dim3(S_ / 16, B_ * H_), 32, 0, stream>>>(qf16, k16, vT16, attn16);

  // 9) out = attn_out @ wo^T (4096 x 2048 x 2048) -> fp32 d_out
  gemm_bf16_nt<ROWS, D_, NO>
      <<<dim3((D_ + 255) / 256, ROWS / 64), 256, 0, stream>>>(attn16, wo16, (float*)d_out);
}
